// MultiHeadAttention_51075751084751
// MI455X (gfx1250) — matrix operand order repaired and checked by reading, not compiled, not run on hardware
//
#include <hip/hip_runtime.h>

// ---------------------------------------------------------------------------
// MI455X (gfx1250) multi-head attention, bf16 WMMA path, fp32 accumulate.
//   b=2, s=2048, d=1024, h=16, dk=64
// Stage 1: QKV = x @ W_qkv + b_qkv   -> Q,K,V bf16 (b,h,s,dk), Q pre-scaled by 1/8
// Stage 2: flash attention per (b,h) -> ctx bf16 (b*s, d)   [K tile via TDM DMA]
// Stage 3: out = ctx @ W_o + b_o     -> fp32
// All GEMM-ish loops are double-buffered in LDS with register-staged prefetch.
// Workspace: 4 buffers of 4,194,304 bf16 = 32 MB total.
// ---------------------------------------------------------------------------

typedef __attribute__((ext_vector_type(16))) __bf16 v16bf;
typedef __attribute__((ext_vector_type(8)))  float  v8f;
typedef __attribute__((ext_vector_type(4)))  unsigned int u32x4;
typedef __attribute__((ext_vector_type(8)))  int          i32x8;
typedef __attribute__((ext_vector_type(4)))  int          i32x4;

#if __has_builtin(__builtin_amdgcn_tensor_load_to_lds)
#define HAVE_TDM 1
#else
#define HAVE_TDM 0
#endif

union AFrag {
    v16bf v;
    uint4 q[2];
};

__device__ __forceinline__ unsigned short f32_bf16(float f) {
    unsigned int u = __float_as_uint(f);
    u += 0x7FFFu + ((u >> 16) & 1u);   // round-to-nearest-even
    return (unsigned short)(u >> 16);
}

__device__ __forceinline__ unsigned int pack2_bf16(float lo, float hi) {
    return (unsigned int)f32_bf16(lo) | ((unsigned int)f32_bf16(hi) << 16);
}

__device__ __forceinline__ v8f wmma_bf16(const AFrag& a, const AFrag& b, v8f c) {
    // D = A(16x32 bf16) * B(32x16 bf16) + C(16x16 f32)
    return __builtin_amdgcn_wmma_f32_16x16x32_bf16(
        false, a.v, false, b.v, (short)0, c, false, false);
}

__device__ __forceinline__ void wait_tensor0() {
#if __has_builtin(__builtin_amdgcn_s_wait_tensorcnt)
    __builtin_amdgcn_s_wait_tensorcnt(0);
#else
    asm volatile("s_wait_tensorcnt 0" ::: "memory");
#endif
}

#if HAVE_TDM
// 1-D contiguous DMA of nelem bf16 elements from global -> LDS via the Tensor
// Data Mover.  D# packed per CDNA5 ISA 8.3/8.4: group0 = {count=1, lds_addr,
// global_addr[56:0], type=2}; group1 = {data_size=1(2B), tensor_dim0=tile_dim0
// =nelem, tensor_dim1=tile_dim1=1, stride0=nelem}.  Tensor==tile -> no OOB.
__device__ __forceinline__ void tdm_load_2b(unsigned int lds_byte_addr,
                                            const void* gptr,
                                            unsigned int nelem) {
    unsigned long long ga = (unsigned long long)gptr;
    u32x4 g0;
    g0[0] = 1u;                                             // count=1
    g0[1] = lds_byte_addr;                                  // lds_addr
    g0[2] = (unsigned int)(ga & 0xFFFFFFFFu);               // global_addr lo
    g0[3] = (unsigned int)((ga >> 32) & 0x01FFFFFFu) | 0x80000000u; // hi + type=2
    i32x8 g1;
    g1[0] = 0x00010000;                                     // data_size=1 (2B)
    g1[1] = (int)((nelem & 0xFFFFu) << 16);                 // tensor_dim0 lo16
    g1[2] = (int)((nelem >> 16) | (1u << 16));              // dim0 hi16 | tensor_dim1=1
    g1[3] = (int)(nelem << 16);                             // tile_dim0
    g1[4] = 1;                                              // tile_dim1=1, tile_dim2=0
    g1[5] = (int)nelem;                                     // tensor_dim0_stride lo32
    g1[6] = 0;
    g1[7] = 0;
    i32x4 z4 = {0, 0, 0, 0};
#if defined(__clang_major__) && (__clang_major__ >= 23)
    i32x8 z8 = {0, 0, 0, 0, 0, 0, 0, 0};
    __builtin_amdgcn_tensor_load_to_lds(g0, g1, z4, z4, z8, 0);
#else
    __builtin_amdgcn_tensor_load_to_lds(g0, g1, z4, z4, 0);
#endif
}
#endif

// ---------------------------------------------------------------------------
// Kernel 1: QKV projection.  M=4096 (b*s), N=3072, K=1024.
// Block tile 128(M) x 64(N), 8 waves, K-step 32, double-buffered LDS.
// ---------------------------------------------------------------------------
__global__ __launch_bounds__(256) void qkv_proj_kernel(
    const float* __restrict__ x, const float* __restrict__ Wqkv,
    const float* __restrict__ bqkv,
    unsigned short* __restrict__ Qo, unsigned short* __restrict__ Ko,
    unsigned short* __restrict__ Vo)
{
    __shared__ __align__(16) unsigned int lA[2][128 * 16];
    __shared__ __align__(16) unsigned int lB[2][64 * 16];

    const int tid  = threadIdx.x;
    const int lane = tid & 31;
    const int wave = tid >> 5;
    const int l16  = lane & 15;
    const int lh   = lane >> 4;
    const int blockN = blockIdx.x * 64;
    const int blockM = blockIdx.y * 128;

    v8f zeroA = {};
    v8f acc[4];
#pragma unroll
    for (int c = 0; c < 4; c++) acc[c] = zeroA;

    float4 fa[4], fb[2];
    auto gload = [&](int ks) {
#pragma unroll
        for (int i = 0; i < 4; i++) {
            int idx = tid + i * 256;
            int row = idx >> 3, q4 = idx & 7;
            fa[i] = *(const float4*)(x + (size_t)(blockM + row) * 1024 + ks + q4 * 4);
        }
        int j = tid & 15, n0 = (tid >> 4) * 4;
        fb[0] = *(const float4*)(Wqkv + (size_t)(ks + 2 * j)     * 3072 + blockN + n0);
        fb[1] = *(const float4*)(Wqkv + (size_t)(ks + 2 * j + 1) * 3072 + blockN + n0);
    };
    auto lstore = [&](unsigned int* A, unsigned int* Bl) {
#pragma unroll
        for (int i = 0; i < 4; i++) {
            int idx = tid + i * 256;
            int row = idx >> 3, q4 = idx & 7;
            uint2 pw;
            pw.x = pack2_bf16(fa[i].x, fa[i].y);
            pw.y = pack2_bf16(fa[i].z, fa[i].w);
            *(uint2*)(&A[row * 16 + q4 * 2]) = pw;
        }
        int j = tid & 15, n0 = (tid >> 4) * 4;
        Bl[(n0 + 0) * 16 + j] = pack2_bf16(fb[0].x, fb[1].x);
        Bl[(n0 + 1) * 16 + j] = pack2_bf16(fb[0].y, fb[1].y);
        Bl[(n0 + 2) * 16 + j] = pack2_bf16(fb[0].z, fb[1].z);
        Bl[(n0 + 3) * 16 + j] = pack2_bf16(fb[0].w, fb[1].w);
    };
    auto compute = [&](const unsigned int* A, const unsigned int* Bl) {
        AFrag a;
        const uint4* ap = (const uint4*)(&A[(wave * 16 + l16) * 16]);
        a.q[0] = ap[2 * lh];
        a.q[1] = ap[2 * lh + 1];
#pragma unroll
        for (int c = 0; c < 4; c++) {
            AFrag b;
            const uint4* bp = (const uint4*)(&Bl[(c * 16 + l16) * 16]);
            b.q[0] = bp[lh ? 2 : 0];
            b.q[1] = bp[lh ? 3 : 1];
            acc[c] = wmma_bf16(a, b, acc[c]);
        }
    };

    gload(0);
    for (int kt = 0; kt < 32; kt++) {
        const int buf = kt & 1;
        lstore(&lA[buf][0], &lB[buf][0]);   // commit tile kt
        if (kt < 31) gload((kt + 1) * 32);  // prefetch tile kt+1 (in flight during compute)
        __syncthreads();
        compute(&lA[buf][0], &lB[buf][0]);
    }

    // Epilogue: bias, fold 1/sqrt(dk)=0.125 into Q, scatter to (b,h,s,dk) bf16.
#pragma unroll
    for (int c = 0; c < 4; c++) {
        const int gcol = blockN + c * 16 + l16;
        const float bias = bqkv[gcol];
        const int head = gcol / 192;
        const int t = gcol - head * 192;
        const int sel = t >> 6;          // 0=Q, 1=K, 2=V
        const int dkidx = t & 63;
        unsigned short* dst = (sel == 0) ? Qo : ((sel == 1) ? Ko : Vo);
        const float scale = (sel == 0) ? 0.125f : 1.0f;
#pragma unroll
        for (int r = 0; r < 8; r++) {
            const int grow = blockM + wave * 16 + r + lh * 8;   // index into (b*s)
            const int bi = grow >> 11;
            const int srow = grow & 2047;
            const float val = (acc[c][r] + bias) * scale;
            dst[(((size_t)bi * 16 + head) * 2048 + srow) * 64 + dkidx] = f32_bf16(val);
        }
    }
}

// ---------------------------------------------------------------------------
// Kernel 2: flash attention per (b,h). grid = (s/64, b*h), block = 128 (4 waves).
// K tile DMA'd by the Tensor Data Mover (wave 0 issues, TENSORcnt fenced),
// V tile repacked to B-layout by VALU; both double-buffered against compute.
// ---------------------------------------------------------------------------
__global__ __launch_bounds__(128) void attn_kernel(
    const unsigned short* __restrict__ Qg, const unsigned short* __restrict__ Kg,
    const unsigned short* __restrict__ Vg, unsigned short* __restrict__ Cg)
{
    __shared__ __align__(16) unsigned short lK[2][32 * 64];   // [key][dk] row-major
    __shared__ __align__(16) unsigned int   lV[2][64 * 16];   // [dk col][k-pair] packed
    __shared__ __align__(16) unsigned short lP[4][16 * 32];   // per-wave P tile

    const int tid  = threadIdx.x;
    const int lane = tid & 31;
    const int wave = tid >> 5;
    const int l16  = lane & 15;
    const int lh   = lane >> 4;
    const int bh   = blockIdx.y;
    const size_t base = (size_t)bh * 2048 * 64;
    const unsigned short* Qb = Qg + base;
    const unsigned short* Kb = Kg + base;
    const unsigned short* Vb = Vg + base;
    const int qblock = blockIdx.x * 64;
    const int qrow = qblock + wave * 16 + l16;

    // Preload Q fragments (two dk-chunks of 32): A-layout direct from global.
    AFrag qa0, qa1;
    const uint4* qp = (const uint4*)(Qb + (size_t)qrow * 64);
    qa0.q[0] = qp[2 * lh];     qa0.q[1] = qp[2 * lh + 1];
    qa1.q[0] = qp[4 + 2 * lh]; qa1.q[1] = qp[5 + 2 * lh];

    // --- staging helpers -----------------------------------------------------
    uint2 va[2], vb[2];
    auto vload = [&](int key0) {
#pragma unroll
        for (int i = 0; i < 2; i++) {
            int unit = tid + i * 128;
            int j = unit & 15, n0 = (unit >> 4) * 4;
            va[i] = *(const uint2*)(Vb + (size_t)(key0 + 2 * j)     * 64 + n0);
            vb[i] = *(const uint2*)(Vb + (size_t)(key0 + 2 * j + 1) * 64 + n0);
        }
    };
    auto vstore = [&](unsigned int* dst) {
#pragma unroll
        for (int i = 0; i < 2; i++) {
            int unit = tid + i * 128;
            int j = unit & 15, n0 = (unit >> 4) * 4;
            dst[(n0 + 0) * 16 + j] = (va[i].x & 0xffffu) | (vb[i].x << 16);
            dst[(n0 + 1) * 16 + j] = (va[i].x >> 16)     | (vb[i].x & 0xffff0000u);
            dst[(n0 + 2) * 16 + j] = (va[i].y & 0xffffu) | (vb[i].y << 16);
            dst[(n0 + 3) * 16 + j] = (va[i].y >> 16)     | (vb[i].y & 0xffff0000u);
        }
    };
#if HAVE_TDM
    auto kprefetch = [&](int key0, int buf) {
        if (wave == 0) {
            unsigned int lds_addr =
                (unsigned int)(unsigned long long)(&lK[buf][0]);
            tdm_load_2b(lds_addr, Kb + (size_t)key0 * 64, 32 * 64);
        }
    };
    auto kcommit = [&](int /*buf*/) {
        if (wave == 0) wait_tensor0();
    };
#else
    uint4 kreg[2];
    auto kprefetch = [&](int key0, int /*buf*/) {
#pragma unroll
        for (int i = 0; i < 2; i++) {
            int idx = tid + i * 128;
            int key = idx >> 3, c8 = idx & 7;
            kreg[i] = *(const uint4*)(Kb + (size_t)(key0 + key) * 64 + c8 * 8);
        }
    };
    auto kcommit = [&](int buf) {
#pragma unroll
        for (int i = 0; i < 2; i++) {
            int idx = tid + i * 128;
            int key = idx >> 3, c8 = idx & 7;
            *(uint4*)(&lK[buf][key * 64 + c8 * 8]) = kreg[i];
        }
    };
#endif

    float mrow[8], lrow[8];
    v8f vzero = {};
    v8f oacc[4];
#pragma unroll
    for (int r = 0; r < 8; r++) { mrow[r] = -1.0e30f; lrow[r] = 0.0f; }
#pragma unroll
    for (int c = 0; c < 4; c++) oacc[c] = vzero;

    // --- prologue: stage tile 0 ---------------------------------------------
    kprefetch(0, 0);
    vload(0);
    vstore(&lV[0][0]);
    kcommit(0);
    __syncthreads();

    for (int kb = 0; kb < 64; kb++) {
        const int cur = kb & 1;
        const int nxt = cur ^ 1;
        if (kb < 63) {
            kprefetch((kb + 1) * 32, nxt);   // TDM writes other buffer during compute
            vload((kb + 1) * 32);
        }

        // Scores: S = Q * K^T (scale pre-folded into Q). Two 16x16 tiles.
        v8f s0 = vzero, s1 = vzero;
        {
            const uint4* k0 = (const uint4*)(&lK[cur][(0 * 16 + l16) * 64]);
            const uint4* k1 = (const uint4*)(&lK[cur][(1 * 16 + l16) * 64]);
            AFrag b00, b01, b10, b11;
            b00.q[0] = k0[lh ? 2 : 0];       b00.q[1] = k0[lh ? 3 : 1];
            b01.q[0] = k0[4 + (lh ? 2 : 0)]; b01.q[1] = k0[4 + (lh ? 3 : 1)];
            b10.q[0] = k1[lh ? 2 : 0];       b10.q[1] = k1[lh ? 3 : 1];
            b11.q[0] = k1[4 + (lh ? 2 : 0)]; b11.q[1] = k1[4 + (lh ? 3 : 1)];
            s0 = wmma_bf16(qa0, b00, s0);
            s0 = wmma_bf16(qa1, b01, s0);
            s1 = wmma_bf16(qa0, b10, s1);
            s1 = wmma_bf16(qa1, b11, s1);
        }

        // Online softmax. Row m lives across 16 lanes of one half-wave in VGPR r.
        unsigned short* pw = lP[wave];
#pragma unroll
        for (int r = 0; r < 8; r++) {
            float v0 = s0[r], v1 = s1[r];
            float mx = fmaxf(v0, v1);
            mx = fmaxf(mx, __shfl_xor(mx, 1, 32));
            mx = fmaxf(mx, __shfl_xor(mx, 2, 32));
            mx = fmaxf(mx, __shfl_xor(mx, 4, 32));
            mx = fmaxf(mx, __shfl_xor(mx, 8, 32));
            float mn = fmaxf(mrow[r], mx);
            float alpha = __expf(mrow[r] - mn);
            float p0 = __expf(v0 - mn);
            float p1 = __expf(v1 - mn);
            float rs = p0 + p1;
            rs += __shfl_xor(rs, 1, 32);
            rs += __shfl_xor(rs, 2, 32);
            rs += __shfl_xor(rs, 4, 32);
            rs += __shfl_xor(rs, 8, 32);
            lrow[r] = lrow[r] * alpha + rs;
            mrow[r] = mn;
#pragma unroll
            for (int c = 0; c < 4; c++) oacc[c][r] *= alpha;
            const int prow = r + lh * 8;
            pw[prow * 32 + l16]      = f32_bf16(p0);
            pw[prow * 32 + 16 + l16] = f32_bf16(p1);
        }
        // DS ops are in-order per wave; fence anyway before A-layout reload of P.
        asm volatile("s_wait_dscnt 0" ::: "memory");

        // ctx: O += P(16x32) * V(32x64)
        AFrag pa;
        const uint4* pp = (const uint4*)(&pw[l16 * 32]);
        pa.q[0] = pp[2 * lh];
        pa.q[1] = pp[2 * lh + 1];
#pragma unroll
        for (int c = 0; c < 4; c++) {
            AFrag vbf;
            const uint4* vp = (const uint4*)(&lV[cur][(c * 16 + l16) * 16]);
            vbf.q[0] = vp[lh ? 2 : 0];
            vbf.q[1] = vp[lh ? 3 : 1];
            oacc[c] = wmma_bf16(pa, vbf, oacc[c]);
        }

        if (kb < 63) {
            vstore(&lV[nxt][0]);   // commit prefetched V tile
            kcommit(nxt);          // wave0: s_wait_tensorcnt 0 for TDM K tile
            __syncthreads();       // publish tile kb+1 to all waves
        }
    }

    // Normalize and write ctx as bf16 in (b*s, d) layout.
    const int bi = bh >> 4, head = bh & 15;
#pragma unroll
    for (int r = 0; r < 8; r++) {
        const float inv = 1.0f / lrow[r];
        const int row = qblock + wave * 16 + r + lh * 8;
        const size_t rb = ((size_t)(bi * 2048 + row)) * 1024 + head * 64;
#pragma unroll
        for (int c = 0; c < 4; c++) {
            Cg[rb + c * 16 + l16] = f32_bf16(oacc[c][r] * inv);
        }
    }
}

// ---------------------------------------------------------------------------
// Kernel 3: output projection. M=4096, N=1024, K=1024. A already bf16.
// ---------------------------------------------------------------------------
__global__ __launch_bounds__(256) void out_proj_kernel(
    const unsigned short* __restrict__ Cg, const float* __restrict__ Wo,
    const float* __restrict__ bo, float* __restrict__ Og)
{
    __shared__ __align__(16) unsigned int lA[2][128 * 16];
    __shared__ __align__(16) unsigned int lB[2][64 * 16];

    const int tid  = threadIdx.x;
    const int lane = tid & 31;
    const int wave = tid >> 5;
    const int l16  = lane & 15;
    const int lh   = lane >> 4;
    const int blockN = blockIdx.x * 64;
    const int blockM = blockIdx.y * 128;

    v8f zeroA = {};
    v8f acc[4];
#pragma unroll
    for (int c = 0; c < 4; c++) acc[c] = zeroA;

    uint4 ua[2];
    float4 fb[2];
    auto gload = [&](int ks) {
#pragma unroll
        for (int i = 0; i < 2; i++) {
            int idx = tid + i * 256;
            int row = idx >> 2, q = idx & 3;
            ua[i] = *(const uint4*)(Cg + (size_t)(blockM + row) * 1024 + ks + q * 8);
        }
        int j = tid & 15, n0 = (tid >> 4) * 4;
        fb[0] = *(const float4*)(Wo + (size_t)(ks + 2 * j)     * 1024 + blockN + n0);
        fb[1] = *(const float4*)(Wo + (size_t)(ks + 2 * j + 1) * 1024 + blockN + n0);
    };
    auto lstore = [&](unsigned int* A, unsigned int* Bl) {
#pragma unroll
        for (int i = 0; i < 2; i++) {
            int idx = tid + i * 256;
            int row = idx >> 2, q = idx & 3;
            *(uint4*)(&A[row * 16 + q * 4]) = ua[i];
        }
        int j = tid & 15, n0 = (tid >> 4) * 4;
        Bl[(n0 + 0) * 16 + j] = pack2_bf16(fb[0].x, fb[1].x);
        Bl[(n0 + 1) * 16 + j] = pack2_bf16(fb[0].y, fb[1].y);
        Bl[(n0 + 2) * 16 + j] = pack2_bf16(fb[0].z, fb[1].z);
        Bl[(n0 + 3) * 16 + j] = pack2_bf16(fb[0].w, fb[1].w);
    };
    auto compute = [&](const unsigned int* A, const unsigned int* Bl) {
        AFrag a;
        const uint4* ap = (const uint4*)(&A[(wave * 16 + l16) * 16]);
        a.q[0] = ap[2 * lh];
        a.q[1] = ap[2 * lh + 1];
#pragma unroll
        for (int c = 0; c < 4; c++) {
            AFrag b;
            const uint4* bp = (const uint4*)(&Bl[(c * 16 + l16) * 16]);
            b.q[0] = bp[lh ? 2 : 0];
            b.q[1] = bp[lh ? 3 : 1];
            acc[c] = wmma_bf16(a, b, acc[c]);
        }
    };

    gload(0);
    for (int kt = 0; kt < 32; kt++) {
        const int buf = kt & 1;
        lstore(&lA[buf][0], &lB[buf][0]);
        if (kt < 31) gload((kt + 1) * 32);
        __syncthreads();
        compute(&lA[buf][0], &lB[buf][0]);
    }

#pragma unroll
    for (int c = 0; c < 4; c++) {
        const int gcol = blockN + c * 16 + l16;
        const float bias = bo[gcol];
#pragma unroll
        for (int r = 0; r < 8; r++) {
            const int grow = blockM + wave * 16 + r + lh * 8;
            Og[(size_t)grow * 1024 + gcol] = acc[c][r] + bias;
        }
    }
}

// ---------------------------------------------------------------------------
extern "C" void kernel_launch(void* const* d_in, const int* in_sizes, int n_in,
                              void* d_out, int out_size, void* d_ws, size_t ws_size,
                              hipStream_t stream) {
    (void)in_sizes; (void)n_in; (void)out_size; (void)ws_size;
    const float* x    = (const float*)d_in[0];
    const float* Wqkv = (const float*)d_in[1];
    const float* bqkv = (const float*)d_in[2];
    const float* Wo   = (const float*)d_in[3];
    const float* bo   = (const float*)d_in[4];
    float* out = (float*)d_out;

    unsigned short* ws = (unsigned short*)d_ws;
    const size_t QSZ = (size_t)2 * 16 * 2048 * 64;  // 4,194,304 bf16 elements
    unsigned short* Qw = ws;
    unsigned short* Kw = ws + QSZ;
    unsigned short* Vw = ws + 2 * QSZ;
    unsigned short* Cw = ws + 3 * QSZ;

    qkv_proj_kernel<<<dim3(48, 32), 256, 0, stream>>>(x, Wqkv, bqkv, Qw, Kw, Vw);
    attn_kernel<<<dim3(32, 32), 128, 0, stream>>>(Qw, Kw, Vw, Cw);
    out_proj_kernel<<<dim3(16, 32), 256, 0, stream>>>(Cw, Wo, bo, out);
}
